// cox_loss_cox_nnet_48249662603869
// MI455X (gfx1250) — matrix operand order repaired and checked by reading, not compiled, not run on hardware
//
#include <hip/hip_runtime.h>
#include <math.h>

// ---------------------------------------------------------------------------
// Cox partial-likelihood loss, fused risk-set reduction via CDNA5 WMMA.
//   risk_sum[i] = sum_j [ytime[j] >= ytime[i]] * exp(theta[j])
//   loss = -mean((theta_i - log(risk_sum_i)) * ystatus_i)
// N = 8192. Mask matrix is never materialized; it is generated per 16x32 tile
// and consumed by v_wmma_f32_16x16x32_f16. exp(theta) is carried as a
// hi+lo f16 split (two chained WMMAs) for ~f32 accuracy.
// ---------------------------------------------------------------------------

typedef _Float16 v16h __attribute__((ext_vector_type(16)));
typedef _Float16 v8h  __attribute__((ext_vector_type(8)));
typedef float    v8f  __attribute__((ext_vector_type(8)));
typedef float    v4f  __attribute__((ext_vector_type(4)));

#define COX_N        8192
#define WAVES        8
#define BLOCK_THREADS (WAVES * 32)
#define ROWS_PER_BLK 16
#define NBLOCKS      (COX_N / ROWS_PER_BLK)          // 512
#define KTILE        32
#define KITERS       (COX_N / (WAVES * KTILE))       // 32

// ---------------------------------------------------------------------------
// Kernel 0: split exp(theta) into hi/lo f16 parts.
// ---------------------------------------------------------------------------
__global__ void cox_prep_kernel(const float* __restrict__ logRR,
                                _Float16* __restrict__ ehi,
                                _Float16* __restrict__ elo) {
    int i = blockIdx.x * blockDim.x + threadIdx.x;
    if (i < COX_N) {
        float e = expf(logRR[i]);
        _Float16 h = (_Float16)e;
        ehi[i] = h;
        elo[i] = (_Float16)(e - (float)h);
    }
}

// ---------------------------------------------------------------------------
// Kernel 1: one block per 16-row i-block. Each of the 8 waves accumulates a
// 16x16 WMMA tile over its share of the 8192-column risk set; waves are
// reduced through LDS; per-block partial term sum -> partials[blockIdx].
// ---------------------------------------------------------------------------
__global__ void __launch_bounds__(BLOCK_THREADS)
cox_main_kernel(const float* __restrict__ logRR,
                const float* __restrict__ ytime,
                const float* __restrict__ ystatus,
                const _Float16* __restrict__ ehi,
                const _Float16* __restrict__ elo,
                float* __restrict__ partials) {
    const int tid  = threadIdx.x;
    const int lane = tid & 31;
    const int wave = tid >> 5;
    const int iblock = blockIdx.x;

    const int hiHalf = (lane >= 16) ? 1 : 0;  // lane half selects K sub-range
    const int n      = lane & 15;             // column (= row index i) in tile

    // Row-time for this lane's output column (same for both lane halves).
    const float yt_i = ytime[iblock * ROWS_PER_BLK + n];

    // A-layout K offsets (16-bit A 16x32, same K order as B):
    //   lanes 0-15 : elems 0..15 -> K=0..15
    //   lanes 16-31: elems 0..15 -> K=16..31
    const int aOff = hiHalf * 16;
    // B-layout K offset (16-bit B 32x16): lanes 0-15 -> K=0..15, 16-31 -> K=16..31
    const int bOff = hiHalf * 16;

    v8f acc = {};  // f32 16x16 accumulator (8 VGPRs)

    for (int it = 0; it < KITERS; ++it) {
        const int jbase = wave * KTILE + it * (WAVES * KTILE);

        // ---- A matrices: exp(theta) hi/lo, identical across the 16 rows ----
        v8h h0 = *(const v8h*)(ehi + jbase + aOff);
        v8h h1 = *(const v8h*)(ehi + jbase + 8 + aOff);
        v8h l0 = *(const v8h*)(elo + jbase + aOff);
        v8h l1 = *(const v8h*)(elo + jbase + 8 + aOff);
        v16h a_hi, a_lo;
#pragma unroll
        for (int e = 0; e < 8; ++e) {
            a_hi[e]     = h0[e];
            a_hi[8 + e] = h1[e];
            a_lo[e]     = l0[e];
            a_lo[8 + e] = l1[e];
        }

        // ---- B matrix: risk-set mask, generated on the fly (exact in f16) ----
        v4f y0 = *(const v4f*)(ytime + jbase + bOff + 0);
        v4f y1 = *(const v4f*)(ytime + jbase + bOff + 4);
        v4f y2 = *(const v4f*)(ytime + jbase + bOff + 8);
        v4f y3 = *(const v4f*)(ytime + jbase + bOff + 12);
        v16h b;
#pragma unroll
        for (int e = 0; e < 4; ++e) {
            b[e]      = (y0[e] >= yt_i) ? (_Float16)1.0f : (_Float16)0.0f;
            b[4 + e]  = (y1[e] >= yt_i) ? (_Float16)1.0f : (_Float16)0.0f;
            b[8 + e]  = (y2[e] >= yt_i) ? (_Float16)1.0f : (_Float16)0.0f;
            b[12 + e] = (y3[e] >= yt_i) ? (_Float16)1.0f : (_Float16)0.0f;
        }

        // ---- Two chained WMMAs: acc += (A_hi + A_lo) x B in f32 ----
        acc = __builtin_amdgcn_wmma_f32_16x16x32_f16(
            false, a_hi, false, b, (short)0, acc, false, false);
        acc = __builtin_amdgcn_wmma_f32_16x16x32_f16(
            false, a_lo, false, b, (short)0, acc, false, false);
    }

    // C layout: VGPR0, lanes 0-15 hold C[M=0][N=lane]; A rows are identical,
    // so row 0 carries the full partial risk sum for column `lane`.
    __shared__ float riskPart[WAVES][ROWS_PER_BLK];
    __shared__ float termArr[ROWS_PER_BLK];

    if (lane < 16) riskPart[wave][lane] = acc[0];
    __syncthreads();

    if (tid < ROWS_PER_BLK) {
        float r = 0.0f;
#pragma unroll
        for (int w = 0; w < WAVES; ++w) r += riskPart[w][tid];
        const int i = iblock * ROWS_PER_BLK + tid;
        termArr[tid] = (logRR[i] - logf(r)) * ystatus[i];
    }
    __syncthreads();

    if (tid == 0) {
        float s = 0.0f;
#pragma unroll
        for (int k = 0; k < ROWS_PER_BLK; ++k) s += termArr[k];
        partials[iblock] = s;
    }
}

// ---------------------------------------------------------------------------
// Kernel 2: deterministic tree reduction of the 512 block partials.
// ---------------------------------------------------------------------------
__global__ void cox_final_kernel(const float* __restrict__ partials,
                                 float* __restrict__ out) {
    __shared__ float s[NBLOCKS];
    const int t = threadIdx.x;
    s[t] = partials[t];
    __syncthreads();
#pragma unroll
    for (int o = NBLOCKS / 2; o > 0; o >>= 1) {
        if (t < o) s[t] += s[t + o];
        __syncthreads();
    }
    if (t == 0) out[0] = -s[0] * (1.0f / (float)COX_N);
}

// ---------------------------------------------------------------------------
extern "C" void kernel_launch(void* const* d_in, const int* in_sizes, int n_in,
                              void* d_out, int out_size, void* d_ws, size_t ws_size,
                              hipStream_t stream) {
    const float* logRR   = (const float*)d_in[0];  // [N,1] f32
    const float* ytime   = (const float*)d_in[1];  // [N]   f32
    const float* ystatus = (const float*)d_in[2];  // [N]   f32
    float* out = (float*)d_out;                    // scalar f32 loss

    char* ws = (char*)d_ws;
    _Float16* ehi     = (_Float16*)(ws);                               // 16 KB
    _Float16* elo     = (_Float16*)(ws + COX_N * sizeof(_Float16));    // 16 KB
    float*    partials = (float*)(ws + 2 * COX_N * sizeof(_Float16));  //  2 KB

    cox_prep_kernel<<<COX_N / 256, 256, 0, stream>>>(logRR, ehi, elo);
    cox_main_kernel<<<NBLOCKS, BLOCK_THREADS, 0, stream>>>(
        logRR, ytime, ystatus, ehi, elo, partials);
    cox_final_kernel<<<1, NBLOCKS, 0, stream>>>(partials, out);
}
